// MHCModuler_9809705304400
// MI455X (gfx1250) — matrix operand order repaired and checked by reading, compiled, not run on hardware
//
#include <hip/hip_runtime.h>
#include <hip/hip_bf16.h>

// MHC moduler for MI455X (gfx1250, wave32).
// Kernel 0: one-shot hc_fn fp32 -> bf16 (shared B operand, 192 KB in d_ws).
// Kernel 1: WMMA bf16 mat-vec (weights) + in-register 4x4 Sinkhorn -> fused 4x4 M per token (d_ws).
// Kernel 2: out[j,h] = sum_i M[i][j] * x[i,h]; x re-read hits the 192MB L2, out stored non-temporally.

typedef __attribute__((ext_vector_type(16))) __bf16 v16bf;
typedef __attribute__((ext_vector_type(4)))  __bf16 v4bf;
typedef __attribute__((ext_vector_type(8)))  float  v8f;
typedef __attribute__((ext_vector_type(4)))  float  v4f;

#define NS        4
#define HDIM      1024
#define DDIM      4096          // NS * HDIM
#define MIXHC     24            // (2+NS)*NS
#define SINK_IT   20
#define HC_EPS    1e-6f
#define NORM_EPS  1e-6f
#define WPB       4             // waves per block, kernel 1
#define BLK1      (WPB * 32)

__device__ __forceinline__ float sigmoidf_(float z) {
    return 1.0f / (1.0f + __expf(-z));
}

// ---------------------------------------------------------------- kernel 0
__global__ __launch_bounds__(256)
void mhc_k0_cvt_b(const float* __restrict__ hc_fn, __bf16* __restrict__ bfn)
{
    const int idx = (blockIdx.x * 256 + threadIdx.x) * 4;   // MIXHC*DDIM total
    v4f v = *(const v4f*)(hc_fn + idx);
    v4bf o;
    o.x = (__bf16)v.x; o.y = (__bf16)v.y; o.z = (__bf16)v.z; o.w = (__bf16)v.w;
    *(v4bf*)(bfn + idx) = o;
}

// ---------------------------------------------------------------- kernel 1
__global__ __launch_bounds__(BLK1)
void mhc_k1_weights(const float* __restrict__ x,
                    const __bf16* __restrict__ bfn,      // hc_fn in bf16
                    const float* __restrict__ hc_scale,
                    const float* __restrict__ hc_base,
                    float* __restrict__ coef)
{
    // per-wave 16x32 weight de-stripe buffer (stride 33 to dodge bank conflicts)
    __shared__ float Wlds[WPB][16][33];

    const int lane = threadIdx.x & 31;
    const int wv   = threadIdx.x >> 5;
    const int tile = blockIdx.x * WPB + wv;   // one 16-token tile per wave
    const int tok0 = tile * 16;
    const int l    = lane & 15;
    const int hi   = lane >> 4;               // half-wave id

    // A operand: lane l / lane l+16 jointly cover row (token) tok0+l
    const float* arow = x + (size_t)(tok0 + l) * DDIM;
    // B operand: lane n reads bf16 hc_fn row n; pad rows 24..31 alias row l (cols ignored)
    const int r1 = (l < MIXHC - 16) ? (16 + l) : l;
    const __bf16* brow0 = bfn + (size_t)l  * DDIM;
    const __bf16* brow1 = bfn + (size_t)r1 * DDIM;

    v8f acc0 = {0.f,0.f,0.f,0.f,0.f,0.f,0.f,0.f};   // weight cols 0..15
    v8f acc1 = {0.f,0.f,0.f,0.f,0.f,0.f,0.f,0.f};   // weight cols 16..31 (16..23 valid)
    float ssq = 0.f;

    const int ao = hi * 16;   // A: K offsets {ao..ao+15}
    const int bo = hi * 16;   // B: K offsets {bo..bo+15}

    #pragma unroll 2
    for (int kk = 0; kk < DDIM; kk += 32) {
        __builtin_prefetch(arow + kk + 256, 0, 3);   // global_prefetch_b8

        v4f a0 = *(const v4f*)(arow + kk + ao);
        v4f a1 = *(const v4f*)(arow + kk + ao + 4);
        v4f a2 = *(const v4f*)(arow + kk + ao + 8);
        v4f a3 = *(const v4f*)(arow + kk + ao + 12);

        ssq += a0.x*a0.x + a0.y*a0.y + a0.z*a0.z + a0.w*a0.w
             + a1.x*a1.x + a1.y*a1.y + a1.z*a1.z + a1.w*a1.w
             + a2.x*a2.x + a2.y*a2.y + a2.z*a2.z + a2.w*a2.w
             + a3.x*a3.x + a3.y*a3.y + a3.z*a3.z + a3.w*a3.w;

        v16bf A;
        A[0]=(__bf16)a0.x;  A[1]=(__bf16)a0.y;  A[2]=(__bf16)a0.z;  A[3]=(__bf16)a0.w;
        A[4]=(__bf16)a1.x;  A[5]=(__bf16)a1.y;  A[6]=(__bf16)a1.z;  A[7]=(__bf16)a1.w;
        A[8]=(__bf16)a2.x;  A[9]=(__bf16)a2.y;  A[10]=(__bf16)a2.z; A[11]=(__bf16)a2.w;
        A[12]=(__bf16)a3.x; A[13]=(__bf16)a3.y; A[14]=(__bf16)a3.z; A[15]=(__bf16)a3.w;

        // B already bf16: 16 consecutive K values = 32 B per lane, no conversion
        v16bf B0 = *(const v16bf*)(brow0 + kk + bo);
        v16bf B1 = *(const v16bf*)(brow1 + kk + bo);

        acc0 = __builtin_amdgcn_wmma_f32_16x16x32_bf16(false, A, false, B0,
                                                       (short)0, acc0, false, false);
        acc1 = __builtin_amdgcn_wmma_f32_16x16x32_bf16(false, A, false, B1,
                                                       (short)0, acc1, false, false);
    }

    // lanes l and l+16 each summed half of token tok0+l's squares
    ssq += __shfl_xor(ssq, 16, 32);
    const float rinv = rsqrtf(ssq * (1.0f / (float)DDIM) + NORM_EPS);

    // de-stripe D-matrix (16x16 f32 C/D layout) into Wlds[wv][token][weight]
    #pragma unroll
    for (int r = 0; r < 8; ++r) {
        Wlds[wv][hi * 8 + r][l]      = acc0[r];
        Wlds[wv][hi * 8 + r][16 + l] = acc1[r];
    }
    __syncthreads();

    if (hi == 0) {
        const int m   = l;               // lane m owns token tok0+m (rinv matches)
        const int tok = tok0 + m;
        const float s0 = hc_scale[0], s1 = hc_scale[1], s2 = hc_scale[2];

        float w[MIXHC];
        #pragma unroll
        for (int n = 0; n < MIXHC; ++n) w[n] = Wlds[wv][m][n] * rinv;

        float hpre[NS], hpost[NS], hres[NS * NS];
        #pragma unroll
        for (int n = 0; n < NS; ++n) {
            hpre[n]  = sigmoidf_(w[n] * s0 + hc_base[n]) + HC_EPS;
            hpost[n] = 2.0f * sigmoidf_(w[NS + n] * s1 + hc_base[NS + n]);
        }
        #pragma unroll
        for (int k = 0; k < NS * NS; ++k)
            hres[k] = w[2 * NS + k] * s2 + hc_base[2 * NS + k];

        // sinkhorn_knopps: row softmax (+eps), col-normalize, then 19 x (row, col)
        #pragma unroll
        for (int i = 0; i < NS; ++i) {
            float mx = fmaxf(fmaxf(hres[i*4+0], hres[i*4+1]),
                             fmaxf(hres[i*4+2], hres[i*4+3]));
            float e0 = __expf(hres[i*4+0] - mx), e1 = __expf(hres[i*4+1] - mx);
            float e2 = __expf(hres[i*4+2] - mx), e3 = __expf(hres[i*4+3] - mx);
            float inv = 1.0f / (e0 + e1 + e2 + e3);
            hres[i*4+0] = e0 * inv + HC_EPS;
            hres[i*4+1] = e1 * inv + HC_EPS;
            hres[i*4+2] = e2 * inv + HC_EPS;
            hres[i*4+3] = e3 * inv + HC_EPS;
        }
        #pragma unroll
        for (int j = 0; j < NS; ++j) {     // first col-normalize
            float inv = 1.0f / (hres[0*4+j] + hres[1*4+j] + hres[2*4+j] + hres[3*4+j] + HC_EPS);
            hres[0*4+j] *= inv; hres[1*4+j] *= inv; hres[2*4+j] *= inv; hres[3*4+j] *= inv;
        }
        for (int it = 0; it < SINK_IT - 1; ++it) {
            #pragma unroll
            for (int i = 0; i < NS; ++i) {
                float inv = 1.0f / (hres[i*4+0] + hres[i*4+1] + hres[i*4+2] + hres[i*4+3] + HC_EPS);
                hres[i*4+0] *= inv; hres[i*4+1] *= inv; hres[i*4+2] *= inv; hres[i*4+3] *= inv;
            }
            #pragma unroll
            for (int j = 0; j < NS; ++j) {
                float inv = 1.0f / (hres[0*4+j] + hres[1*4+j] + hres[2*4+j] + hres[3*4+j] + HC_EPS);
                hres[0*4+j] *= inv; hres[1*4+j] *= inv; hres[2*4+j] *= inv; hres[3*4+j] *= inv;
            }
        }

        // fused combine matrix: M[i][j] = h_post[j]*h_pre[i] + h_res[i][j]
        v4f* cp = (v4f*)(coef + (size_t)tok * 16);
        #pragma unroll
        for (int i = 0; i < NS; ++i) {
            v4f mv;
            mv.x = hpost[0] * hpre[i] + hres[i*4+0];
            mv.y = hpost[1] * hpre[i] + hres[i*4+1];
            mv.z = hpost[2] * hpre[i] + hres[i*4+2];
            mv.w = hpost[3] * hpre[i] + hres[i*4+3];
            cp[i] = mv;
        }
    }
}

// ---------------------------------------------------------------- kernel 2
__global__ __launch_bounds__(256)
void mhc_k2_combine(const float* __restrict__ x,
                    const float* __restrict__ coef,
                    float* __restrict__ out)
{
    const int tok = blockIdx.x;          // one token per block
    const int t   = threadIdx.x;         // 256 threads x float4 = HDIM
    const float* xr = x + (size_t)tok * DDIM;

    v4f xi[NS];
    #pragma unroll
    for (int i = 0; i < NS; ++i)
        xi[i] = ((const v4f*)(xr + i * HDIM))[t];

    const float* Mt = coef + (size_t)tok * 16;   // uniform -> scalar loads
    float M[16];
    #pragma unroll
    for (int k = 0; k < 16; ++k) M[k] = Mt[k];

    float* orow = out + (size_t)tok * DDIM;
    #pragma unroll
    for (int j = 0; j < NS; ++j) {
        v4f o;
        o.x = M[0*4+j]*xi[0].x + M[1*4+j]*xi[1].x + M[2*4+j]*xi[2].x + M[3*4+j]*xi[3].x;
        o.y = M[0*4+j]*xi[0].y + M[1*4+j]*xi[1].y + M[2*4+j]*xi[2].y + M[3*4+j]*xi[3].y;
        o.z = M[0*4+j]*xi[0].z + M[1*4+j]*xi[1].z + M[2*4+j]*xi[2].z + M[3*4+j]*xi[3].z;
        o.w = M[0*4+j]*xi[0].w + M[1*4+j]*xi[1].w + M[2*4+j]*xi[2].w + M[3*4+j]*xi[3].w;
        // non-temporal store: don't evict the L2-resident x stream with out
        __builtin_nontemporal_store(o, (v4f*)(orow + j * HDIM) + t);
    }
}

extern "C" void kernel_launch(void* const* d_in, const int* in_sizes, int n_in,
                              void* d_out, int out_size, void* d_ws, size_t ws_size,
                              hipStream_t stream) {
    const float* x        = (const float*)d_in[0];   // (B,S,ns,H) f32
    const float* hc_fn    = (const float*)d_in[1];   // (24, 4096) f32
    const float* hc_scale = (const float*)d_in[2];   // (3,) f32
    const float* hc_base  = (const float*)d_in[3];   // (24,) f32
    float* out  = (float*)d_out;

    const int ntok  = in_sizes[0] / DDIM;            // 8192
    const int tiles = ntok / 16;                     // 512 (16 tokens / wave)

    // workspace layout: [coef: ntok*16 f32 = 512 KB][bfn: MIXHC*DDIM bf16 = 192 KB]
    float*  coef = (float*)d_ws;
    __bf16* bfn  = (__bf16*)((char*)d_ws + (size_t)ntok * 16 * sizeof(float));

    mhc_k0_cvt_b<<<(MIXHC * DDIM) / (256 * 4), 256, 0, stream>>>(hc_fn, bfn);
    mhc_k1_weights<<<tiles / WPB, BLK1, 0, stream>>>(x, bfn, hc_scale, hc_base, coef);
    mhc_k2_combine<<<ntok, 256, 0, stream>>>(x, coef, out);
}
